// GIN_35613868819113
// MI455X (gfx1250) — matrix operand order repaired and checked by reading, not compiled, not run on hardware
//
#include <hip/hip_runtime.h>

// GIN forward on gfx1250 (MI455X): f32 scatter-agg (L2-resident atomics),
// bf16 WMMA GEMMs with f32 accumulate, f32 BN(train)+ReLU, pooled classifier.
//
// v2: pre-swizzled bf16 weight fragments (coalesced b128 loads), bf16 A tile
//     staged in LDS (convert once, ds_load_b128 fragments), 32-row blocks
//     (8 WMMAs/wave, B fragments register-resident across M-subtiles).

#define N_NODES  100000
#define N_EDGES  3200000
#define HID      128
#define OUT_CH   16
#define N_GRAPHS 512
#define BN_EPS   1e-5f

typedef __bf16 bf16_t;
typedef bf16_t v16bf __attribute__((ext_vector_type(16)));
typedef bf16_t v8bf  __attribute__((ext_vector_type(8)));
typedef float  v8f   __attribute__((ext_vector_type(8)));

// ---------------- small utility kernels ----------------

__global__ void zero_f32(float* __restrict__ p, int n) {
  int i = blockIdx.x * blockDim.x + threadIdx.x;
  for (; i < n; i += gridDim.x * blockDim.x) p[i] = 0.0f;
}

__global__ void copy_f4(const float4* __restrict__ s, float4* __restrict__ d, int n4) {
  int i = blockIdx.x * blockDim.x + threadIdx.x;
  for (; i < n4; i += gridDim.x * blockDim.x) d[i] = s[i];
}

// Pre-swizzle W (f32, K-major [128 x 128]) into per-lane WMMA B fragments:
//   packed[((nt*4 + ks)*32 + lane)*16 + i] = bf16( W[(ks*32 + (i>>3)*16 + (lane>>4)*8 + (i&7))*128
//                                                    + nt*16 + (lane&15)] )
// so each lane's 16 B values for a (ntile, kslab) are 32 contiguous bytes.
__global__ void pack_w_frag(const float* __restrict__ W, bf16_t* __restrict__ out) {
  int idx = blockIdx.x * blockDim.x + threadIdx.x;  // 0 .. 16383
  int i    = idx & 15;
  int lane = (idx >> 4) & 31;
  int ks   = (idx >> 9) & 3;
  int nt   = idx >> 11;
  int n = nt * 16 + (lane & 15);
  int k = ks * 32 + (i >> 3) * 16 + (lane >> 4) * 8 + (i & 7);
  out[idx] = (bf16_t)W[k * HID + n];
}

// ---------------- edge scatter-add: agg[dst] += x[src] ----------------
// one wave per edge; lane handles 4 consecutive features (float4 gather,
// 4x global_atomic_add_f32 scatter). x and agg are L2-resident (51.2MB each).

__global__ void __launch_bounds__(256)
edge_scatter(const float* __restrict__ x, float* __restrict__ agg,
             const long long* __restrict__ ei) {
  int e = blockIdx.x * 8 + (threadIdx.x >> 5);
  if (e >= N_EDGES) return;
  int lane = threadIdx.x & 31;
  long long s = ei[e];
  long long d = ei[(long long)N_EDGES + e];
  float4 v = *(const float4*)(x + s * HID + lane * 4);
  float* p = agg + d * HID + lane * 4;
  atomicAdd(p + 0, v.x);
  atomicAdd(p + 1, v.y);
  atomicAdd(p + 2, v.z);
  atomicAdd(p + 3, v.w);
}

// ---------------- WMMA GEMM: out[R x 128] = act(A[R x 128] @ W[128 x 128] + b) ---
// Block = 256 threads = 8 waves; block tile = 32 rows x 128 cols.
// Wave w owns columns [16w, 16w+16) for both 16-row subtiles:
// 2 subtiles x 4 k-slabs = 8 chained v_wmma_f32_16x16x32_bf16 per wave,
// with the 4 B fragments held in registers across subtiles.
//
// A fragment (ISA 7.12.2, 16-bit A 16x32): lane m=lane&15, khalf=lane>>4;
//   elements 0..7  -> k = ks*32 + khalf*8 + i        (contiguous 8 bf16)
//   elements 8..15 -> k = ks*32 + 16 + khalf*8 + i-8 (contiguous 8 bf16)
// -> two ds_load_b128 from the bf16 LDS tile.
// C/D (32-bit 16x16): VGPR r -> row r + 8*(lane>>4), col lane&15.

__global__ void __launch_bounds__(256)
gemm128_bf16(const float* __restrict__ A, const bf16_t* __restrict__ Wp,
             const float* __restrict__ bias, float* __restrict__ out, int doRelu) {
  __shared__ bf16_t ldsA[32 * 128];  // 8 KB, bf16 row-major tile

  const int rowBase = blockIdx.x * 32;

  // stage + convert: 32x128 f32 -> bf16, each thread 16 consecutive elements
  {
    const float4* src = (const float4*)(A + (size_t)rowBase * HID);
    int t = threadIdx.x;
    float4 f0 = src[t * 4 + 0];
    float4 f1 = src[t * 4 + 1];
    float4 f2 = src[t * 4 + 2];
    float4 f3 = src[t * 4 + 3];
    v8bf p0, p1;
    p0[0] = (bf16_t)f0.x; p0[1] = (bf16_t)f0.y; p0[2] = (bf16_t)f0.z; p0[3] = (bf16_t)f0.w;
    p0[4] = (bf16_t)f1.x; p0[5] = (bf16_t)f1.y; p0[6] = (bf16_t)f1.z; p0[7] = (bf16_t)f1.w;
    p1[0] = (bf16_t)f2.x; p1[1] = (bf16_t)f2.y; p1[2] = (bf16_t)f2.z; p1[3] = (bf16_t)f2.w;
    p1[4] = (bf16_t)f3.x; p1[5] = (bf16_t)f3.y; p1[6] = (bf16_t)f3.z; p1[7] = (bf16_t)f3.w;
    *(v8bf*)(ldsA + t * 16)     = p0;
    *(v8bf*)(ldsA + t * 16 + 8) = p1;
  }
  __syncthreads();

  const int lane  = threadIdx.x & 31;
  const int wave  = threadIdx.x >> 5;   // 0..7 -> N tile
  const int m     = lane & 15;
  const int khalf = lane >> 4;
  const int ncol  = wave * 16 + m;      // B/C column index for this lane

  // B fragments: 4 k-slabs, 32 contiguous bytes per lane, L2/WGP$-resident
  v16bf bfr[4];
#pragma unroll
  for (int ks = 0; ks < 4; ++ks)
    bfr[ks] = *(const v16bf*)(Wp + (size_t)((wave * 4 + ks) * 32 + lane) * 16);

  v8f c0 = {}, c1 = {};
#pragma unroll
  for (int ks = 0; ks < 4; ++ks) {
    const bf16_t* ar = ldsA + m * 128 + ks * 32 + khalf * 8;
    v8bf lo = *(const v8bf*)(ar);
    v8bf hi = *(const v8bf*)(ar + 16);
    v16bf a = __builtin_shufflevector(lo, hi, 0, 1, 2, 3, 4, 5, 6, 7,
                                      8, 9, 10, 11, 12, 13, 14, 15);
    c0 = __builtin_amdgcn_wmma_f32_16x16x32_bf16(false, a, false, bfr[ks],
                                                 (short)0, c0, false, false);
  }
#pragma unroll
  for (int ks = 0; ks < 4; ++ks) {
    const bf16_t* ar = ldsA + (16 + m) * 128 + ks * 32 + khalf * 8;
    v8bf lo = *(const v8bf*)(ar);
    v8bf hi = *(const v8bf*)(ar + 16);
    v16bf a = __builtin_shufflevector(lo, hi, 0, 1, 2, 3, 4, 5, 6, 7,
                                      8, 9, 10, 11, 12, 13, 14, 15);
    c1 = __builtin_amdgcn_wmma_f32_16x16x32_bf16(false, a, false, bfr[ks],
                                                 (short)0, c1, false, false);
  }

  const int mrow = khalf * 8;
  const float bv = bias[ncol];
#pragma unroll
  for (int r = 0; r < 8; ++r) {
    float v = c0[r] + bv;
    if (doRelu) v = fmaxf(v, 0.0f);
    out[(size_t)(rowBase + mrow + r) * HID + ncol] = v;
  }
#pragma unroll
  for (int r = 0; r < 8; ++r) {
    float v = c1[r] + bv;
    if (doRelu) v = fmaxf(v, 0.0f);
    out[(size_t)(rowBase + 16 + mrow + r) * HID + ncol] = v;
  }
}

// ---------------- BatchNorm (training stats) ----------------

__global__ void __launch_bounds__(256)
bn_stats(const float* __restrict__ h, float* __restrict__ sum,
         float* __restrict__ sumsq) {
  int f = threadIdx.x & 127;
  int r = blockIdx.x * 2 + (threadIdx.x >> 7);
  float s = 0.0f, q = 0.0f;
  for (; r < N_NODES; r += gridDim.x * 2) {
    float v = h[(size_t)r * HID + f];
    s += v;
    q += v * v;
  }
  atomicAdd(&sum[f], s);
  atomicAdd(&sumsq[f], q);
}

__global__ void __launch_bounds__(256)
bn_apply_relu(const float* __restrict__ h, const float* __restrict__ sum,
              const float* __restrict__ sumsq, const float* __restrict__ gamma,
              const float* __restrict__ beta, float* __restrict__ dst1,
              float* __restrict__ dst2) {
  const float invN = 1.0f / (float)N_NODES;
  int i = blockIdx.x * blockDim.x + threadIdx.x;
  const int total = N_NODES * HID;
  for (; i < total; i += gridDim.x * blockDim.x) {
    int f = i & 127;
    float mu  = sum[f] * invN;
    float var = sumsq[f] * invN - mu * mu;   // biased variance, matches ref
    float v = (h[i] - mu) * rsqrtf(var + BN_EPS) * gamma[f] + beta[f];
    v = fmaxf(v, 0.0f);
    dst1[i] = v;
    if (dst2) dst2[i] = v;                   // agg init copy for next conv
  }
}

// ---------------- global_add_pool ----------------

__global__ void __launch_bounds__(256)
pool_sum(const float* __restrict__ h, const long long* __restrict__ batch,
         float* __restrict__ pooled) {
  int node = blockIdx.x * 8 + (threadIdx.x >> 5);
  if (node >= N_NODES) return;
  int lane = threadIdx.x & 31;
  long long g = batch[node];
  float4 v = *(const float4*)(h + (size_t)node * HID + lane * 4);
  float* p = pooled + g * HID + lane * 4;
  atomicAdd(p + 0, v.x);
  atomicAdd(p + 1, v.y);
  atomicAdd(p + 2, v.z);
  atomicAdd(p + 3, v.w);
}

// ---------------- classifier: out[512x16] = pooled @ Wlin + blin --------

__global__ void __launch_bounds__(256)
final_lin(const float* __restrict__ pooled, const float* __restrict__ W,
          const float* __restrict__ b, float* __restrict__ out) {
  int i = blockIdx.x * blockDim.x + threadIdx.x;  // 0..8191
  int g = i >> 4;
  int o = i & 15;
  float acc = b[o];
  const float* pr = pooled + g * HID;
#pragma unroll 8
  for (int k = 0; k < HID; ++k) acc = fmaf(pr[k], W[k * OUT_CH + o], acc);
  out[i] = acc;
}

// ---------------- launcher ----------------

extern "C" void kernel_launch(void* const* d_in, const int* in_sizes, int n_in,
                              void* d_out, int out_size, void* d_ws, size_t ws_size,
                              hipStream_t stream) {
  (void)in_sizes; (void)n_in; (void)out_size; (void)ws_size;

  const float*     x     = (const float*)d_in[0];
  const long long* ei    = (const long long*)d_in[1];
  const long long* batch = (const long long*)d_in[2];
  const float* W1a = (const float*)d_in[3];
  const float* b1a = (const float*)d_in[4];
  const float* W1b = (const float*)d_in[5];
  const float* b1b = (const float*)d_in[6];
  const float* g1  = (const float*)d_in[7];
  const float* be1 = (const float*)d_in[8];
  const float* W2a = (const float*)d_in[9];
  const float* b2a = (const float*)d_in[10];
  const float* W2b = (const float*)d_in[11];
  const float* b2b = (const float*)d_in[12];
  const float* g2  = (const float*)d_in[13];
  const float* be2 = (const float*)d_in[14];
  const float* Wl  = (const float*)d_in[15];
  const float* bl  = (const float*)d_in[16];
  float* out = (float*)d_out;

  // workspace layout (~154 MB): 3 ping-pong node buffers + packed weights + stats
  char* ws = (char*)d_ws;
  const size_t S = (size_t)N_NODES * HID * sizeof(float);  // 51.2 MB
  float*  bufA  = (float*)(ws);
  float*  bufB  = (float*)(ws + S);
  float*  bufC  = (float*)(ws + 2 * S);
  bf16_t* wb1a  = (bf16_t*)(ws + 3 * S);
  bf16_t* wb1b  = wb1a + HID * HID;
  bf16_t* wb2a  = wb1b + HID * HID;
  bf16_t* wb2b  = wb2a + HID * HID;
  float*  stats = (float*)(wb2b + HID * HID);  // sum[128] | sumsq[128]
  float*  pooled = stats + 256;                // 512*128 f32

  // weights -> pre-swizzled bf16 WMMA B fragments (L2/WGP$-resident)
  pack_w_frag<<<64, 256, 0, stream>>>(W1a, wb1a);
  pack_w_frag<<<64, 256, 0, stream>>>(W1b, wb1b);
  pack_w_frag<<<64, 256, 0, stream>>>(W2a, wb2a);
  pack_w_frag<<<64, 256, 0, stream>>>(W2b, wb2b);

  // ---- conv1: hpre = x + scatter_add(x[src] -> dst) ----
  copy_f4<<<12500, 256, 0, stream>>>((const float4*)x, (float4*)bufA, N_NODES * HID / 4);
  edge_scatter<<<N_EDGES / 8, 256, 0, stream>>>(x, bufA, ei);
  gemm128_bf16<<<N_NODES / 32, 256, 0, stream>>>(bufA, wb1a, b1a, bufB, 1);  // relu
  gemm128_bf16<<<N_NODES / 32, 256, 0, stream>>>(bufB, wb1b, b1b, bufC, 0);
  zero_f32<<<1, 256, 0, stream>>>(stats, 256);
  bn_stats<<<512, 256, 0, stream>>>(bufC, stats, stats + 128);
  bn_apply_relu<<<2048, 256, 0, stream>>>(bufC, stats, stats + 128, g1, be1,
                                          bufA, bufB);  // bufA=h_norm, bufB=agg init

  // ---- conv2 ----
  edge_scatter<<<N_EDGES / 8, 256, 0, stream>>>(bufA, bufB, ei);  // bufB = hpre2
  gemm128_bf16<<<N_NODES / 32, 256, 0, stream>>>(bufB, wb2a, b2a, bufC, 1);  // relu
  gemm128_bf16<<<N_NODES / 32, 256, 0, stream>>>(bufC, wb2b, b2b, bufA, 0);
  zero_f32<<<1, 256, 0, stream>>>(stats, 256);
  bn_stats<<<512, 256, 0, stream>>>(bufA, stats, stats + 128);
  bn_apply_relu<<<2048, 256, 0, stream>>>(bufA, stats, stats + 128, g2, be2,
                                          bufB, nullptr);  // bufB = h_norm2

  // ---- pooling + classifier ----
  zero_f32<<<64, 256, 0, stream>>>(pooled, N_GRAPHS * HID);
  pool_sum<<<N_NODES / 8, 256, 0, stream>>>(bufB, batch, pooled);
  final_lin<<<(N_GRAPHS * OUT_CH) / 256, 256, 0, stream>>>(pooled, Wl, bl, out);
}
